// FMGCNCell_6657199308828
// MI455X (gfx1250) — matrix operand order repaired and checked by reading, not compiled, not run on hardware
//
#include <hip/hip_runtime.h>
#include <hip/hip_bf16.h>
#include <math.h>

// AGCRN cell for MI455X (gfx1250, wave32, WMMA f16 16x16x32).
// Big math: A(2000x2000) @ X(2000x4224) twice -> v_wmma_f32_16x16x32_f16
//           per-node 64x160x{128,64} GEMMs    -> v_wmma_f32_16x16x32_f16
// f16 operands / f32 accumulate; all hot-loop fragment loads are b128
// (global or ds) against 16B-aligned zero-padded strides (NPAD=2016).
// The big GEMM double-buffers its B tile in LDS via
// GLOBAL_LOAD_ASYNC_TO_LDS_B128 (ASYNCcnt), overlapping DMA with WMMA.

typedef __attribute__((ext_vector_type(16))) _Float16 v16h;
typedef __attribute__((ext_vector_type(8)))  _Float16 v8h;
typedef __attribute__((ext_vector_type(8)))  float    v8f;
typedef __attribute__((ext_vector_type(4)))  float    v4f;
typedef __attribute__((ext_vector_type(4)))  int      v4i;

#define NN   2000          // nodes
#define NPAD 2016          // padded k/row stride (mult of 32, 16B-aligned)
#define BB   64            // batch
#define EE   16            // embed dim
#define CAT  66            // DIN + H
#define KI   132           // CHEB_K * CAT
#define KPAD 160           // KI padded to multiple of 32
#define COLS (BB * CAT)    // 4224
#define OG   128           // gate output (2H)
#define OU   64            // update output (H)

#if defined(__AMDGCN__) && __has_builtin(__builtin_amdgcn_global_load_async_to_lds_b128)
#define HAVE_ASYNC_LDS 1
typedef __attribute__((address_space(1))) v4i* gptr_v4i;
typedef __attribute__((address_space(3))) v4i* lptr_v4i;
#else
#define HAVE_ASYNC_LDS 0
#endif

__device__ __forceinline__ void async_wait_all() {
#if HAVE_ASYNC_LDS
#if __has_builtin(__builtin_amdgcn_s_wait_asynccnt)
  __builtin_amdgcn_s_wait_asynccnt(0);
#else
  asm volatile("s_wait_asynccnt 0" ::: "memory");
#endif
#endif
}

__device__ __forceinline__ v8f wmma_f16(v16h a, v16h b, v8f c) {
  return __builtin_amdgcn_wmma_f32_16x16x32_f16(false, a, false, b, (short)0, c,
                                                false, false);
}

// ---- 1. logits = relu(E @ E^T); one wave per 16x16 tile, K=16 padded to 32 ----
__global__ __launch_bounds__(32)
void logits_kernel(const float* __restrict__ emb, float* __restrict__ logits) {
  const int lane = threadIdx.x & 31;
  const int m = lane & 15;
  const int hi = lane >> 4;
  const int i0 = blockIdx.x * 16;
  const int j0 = blockIdx.y * 16;
  // A: lanes<16 hold k=0..7 / lanes>=16 hold k=8..15 (k>=16 is zero padding)
  const v4f* ap = (const v4f*)(emb + (i0 + m) * EE + (hi ? 8 : 0));
  v4f av0 = ap[0], av1 = ap[1];
  v16h a;
#pragma unroll
  for (int e = 0; e < 4; ++e) {
    a[e] = (_Float16)av0[e];
    a[e + 4] = (_Float16)av1[e];
    a[e + 8] = (_Float16)0.f;
    a[e + 12] = (_Float16)0.f;
  }
  // B: lanes<16 hold k=0..7 / lanes>=16 hold k=8..15 (k>=16 is zero padding)
  const v4f* bp = (const v4f*)(emb + (j0 + m) * EE + (hi ? 8 : 0));
  v4f bv0 = bp[0], bv1 = bp[1];
  v16h b;
#pragma unroll
  for (int e = 0; e < 4; ++e) {
    b[e] = (_Float16)bv0[e];
    b[e + 4] = (_Float16)bv1[e];
    b[e + 8] = (_Float16)0.f;
    b[e + 12] = (_Float16)0.f;
  }
  v8f c = {};
  c = wmma_f16(a, b, c);
  const int mofs = hi ? 8 : 0;
#pragma unroll
  for (int r = 0; r < 8; ++r) {
    float v = c[r];
    logits[(size_t)(i0 + r + mofs) * NN + (j0 + m)] = v > 0.f ? v : 0.f;
  }
}

// ---- 2. row softmax -> A as f16 with zero-padded row stride NPAD ----
__global__ __launch_bounds__(256)
void softmax_kernel(const float* __restrict__ logits, _Float16* __restrict__ Ah) {
  __shared__ float red[256];
  const int row = blockIdx.x;
  const int tid = threadIdx.x;
  const float* rp = logits + (size_t)row * NN;
  float mx = -1e30f;
  for (int j = tid; j < NN; j += 256) mx = fmaxf(mx, rp[j]);
  red[tid] = mx; __syncthreads();
  for (int s = 128; s > 0; s >>= 1) {
    if (tid < s) red[tid] = fmaxf(red[tid], red[tid + s]);
    __syncthreads();
  }
  mx = red[0]; __syncthreads();
  float sum = 0.f;
  for (int j = tid; j < NN; j += 256) sum += __expf(rp[j] - mx);
  red[tid] = sum; __syncthreads();
  for (int s = 128; s > 0; s >>= 1) {
    if (tid < s) red[tid] += red[tid + s];
    __syncthreads();
  }
  const float inv = 1.f / red[0];
  for (int j = tid; j < NPAD; j += 256)
    Ah[(size_t)row * NPAD + j] =
        (j < NN) ? (_Float16)(__expf(rp[j] - mx) * inv) : (_Float16)0.f;
}

// ---- 3. pack xin = concat(x, [z*]state), column-major f16, rows padded ----
__global__ __launch_bounds__(256)
void pack_xin_kernel(const float* __restrict__ x, const float* __restrict__ state,
                     const _Float16* __restrict__ zr, int gated,
                     _Float16* __restrict__ XinT) {
  size_t idx = (size_t)blockIdx.x * 256 + threadIdx.x;
  if (idx >= (size_t)COLS * NPAD) return;
  int n = (int)(idx % NPAD);
  int c = (int)(idx / NPAD);
  float v = 0.f;
  if (n < NN) {
    int b = c / CAT, i = c % CAT;
    if (i < 2) {
      v = x[((size_t)b * NN + n) * 2 + i];
    } else {
      float st = state[((size_t)b * NN + n) * OU + (i - 2)];
      if (gated) {
        float z = (float)zr[((size_t)b * NN + n) * OG + (i - 2)];
        v = z * st;
      } else {
        v = st;
      }
    }
  }
  XinT[idx] = (_Float16)v;
}

// ---- 4. C = A @ Xin : 2000 x 4224 x 2000 ------------------------------------
// block = 160 threads = 5 waves -> 5 M-tiles (25 blocks * 5 = 125 exactly).
// B tile (32k x 64col, 4KB) double-buffered in LDS: async copy of chunk k+1
// overlaps with the 4 WMMAs of chunk k; one s_wait_asynccnt + barrier per k.
__global__ __launch_bounds__(160)
void gemm_ax_kernel(const _Float16* __restrict__ Ah,
                    const _Float16* __restrict__ XinT,
                    _Float16* __restrict__ CT) {
  __shared__ __align__(16) _Float16 sB[2][64 * 32];
  const int tid = threadIdx.x;
  const int lane = tid & 31;
  const int wave = tid >> 5;
  const int m = lane & 15;
  const int hi = lane >> 4;
  const int mBase = (blockIdx.x * 5 + wave) * 16;
  const int nBase = blockIdx.y * 64;
  const _Float16* arow = Ah + (size_t)(mBase + m) * NPAD + (hi ? 8 : 0);

  auto stage = [&](int kc, int buf) {
    if (tid < 128) {
#pragma unroll
      for (int j = 0; j < 2; ++j) {
        int q = tid * 2 + j;          // 0..255
        int col = q >> 2, seg = q & 3;
        const _Float16* src =
            XinT + (size_t)(nBase + col) * NPAD + kc * 32 + seg * 8;
        _Float16* dst = &sB[buf][col * 32 + seg * 8];
#if HAVE_ASYNC_LDS
        __builtin_amdgcn_global_load_async_to_lds_b128((gptr_v4i)src,
                                                       (lptr_v4i)dst, 0, 0);
#else
        *(v8h*)dst = *(const v8h*)src;
#endif
      }
      async_wait_all();  // only this wave's outstanding asyncs
    }
  };

  v8f acc[4] = {};
  const int KC = NPAD / 32;  // 63; padded region is zeros
  stage(0, 0);
  __syncthreads();
  for (int kc = 0; kc < KC; ++kc) {
    const int cur = kc & 1;
    // prefetch next chunk into the other buffer (overlaps with WMMA below)
    if (kc + 1 < KC) {
      if (tid < 128) {
#pragma unroll
        for (int j = 0; j < 2; ++j) {
          int q = tid * 2 + j;
          int col = q >> 2, seg = q & 3;
          const _Float16* src =
              XinT + (size_t)(nBase + col) * NPAD + (kc + 1) * 32 + seg * 8;
          _Float16* dst = &sB[cur ^ 1][col * 32 + seg * 8];
#if HAVE_ASYNC_LDS
          __builtin_amdgcn_global_load_async_to_lds_b128((gptr_v4i)src,
                                                         (lptr_v4i)dst, 0, 0);
#else
          *(v8h*)dst = *(const v8h*)src;
#endif
        }
      }
    }
    // A fragment: two contiguous b128 runs per lane
    const int k0 = kc * 32;
    v8h a0 = *(const v8h*)(arow + k0);
    v8h a1 = *(const v8h*)(arow + k0 + 16);
    v16h a;
#pragma unroll
    for (int e = 0; e < 8; ++e) { a[e] = a0[e]; a[e + 8] = a1[e]; }
#pragma unroll
    for (int t = 0; t < 4; ++t) {
      const _Float16* pb = &sB[cur][(t * 16 + m) * 32 + (hi ? 8 : 0)];
      v8h b0 = *(const v8h*)pb;
      v8h b1 = *(const v8h*)(pb + 16);
      v16h b;
#pragma unroll
      for (int e = 0; e < 8; ++e) { b[e] = b0[e]; b[e + 8] = b1[e]; }
      acc[t] = wmma_f16(a, b, acc[t]);
    }
    if (kc + 1 < KC) {
      if (tid < 128) async_wait_all();  // wave-uniform (waves 0..3)
      __syncthreads();
    }
  }
#pragma unroll
  for (int t = 0; t < 4; ++t) {
    v8h hv;
#pragma unroll
    for (int r = 0; r < 8; ++r) hv[r] = (_Float16)acc[t][r];
    *(v8h*)(CT + (size_t)(nBase + t * 16 + m) * NPAD + mBase + (hi ? 8 : 0)) = hv;
  }
}

// ---- 5. per-node weights, transposed to [n][o][KPAD] (zero k-padding) -------
__global__ __launch_bounds__(256)
void node_weights_kernel(const float* __restrict__ emb,
                         const float* __restrict__ wp, int O,
                         _Float16* __restrict__ Wt) {
  size_t idx = (size_t)blockIdx.x * 256 + threadIdx.x;
  size_t total = (size_t)NN * O * KPAD;
  if (idx >= total) return;
  int k = (int)(idx % KPAD);
  size_t rest = idx / KPAD;
  int o = (int)(rest % O);
  int n = (int)(rest / O);
  float s = 0.f;
  if (k < KI) {
#pragma unroll
    for (int d = 0; d < EE; ++d)
      s += emb[n * EE + d] * wp[((size_t)d * KI + k) * O + o];
  }
  Wt[idx] = (_Float16)s;
}

__global__ __launch_bounds__(256)
void node_bias_kernel(const float* __restrict__ emb,
                      const float* __restrict__ bp, int O,
                      float* __restrict__ bias) {
  int idx = blockIdx.x * 256 + threadIdx.x;
  if (idx >= NN * O) return;
  int o = idx % O, n = idx / O;
  float s = 0.f;
#pragma unroll
  for (int d = 0; d < EE; ++d) s += emb[n * EE + d] * bp[d * O + o];
  bias[idx] = s;
}

// ---- 6. per-node GEMM (64 x KPAD x O) + GRU epilogue; 1 block/node ----------
__global__ __launch_bounds__(256)
void node_gemm_kernel(const _Float16* __restrict__ XinT,
                      const _Float16* __restrict__ CT,
                      const _Float16* __restrict__ Wt,
                      const float* __restrict__ bias,
                      const float* __restrict__ state,
                      _Float16* __restrict__ zr, float* __restrict__ out,
                      int O, int mode) {
  __shared__ __align__(16) _Float16 sx[BB * KPAD];  // 64 x 160 f16 = 20 KB
  const int n = blockIdx.x;
  const int tid = threadIdx.x;
  for (int idx = tid; idx < BB * KPAD; idx += 256) {
    int bb = idx / KPAD, c = idx % KPAD;
    _Float16 v = (_Float16)0.f;
    if (c < CAT)     v = XinT[(size_t)(bb * CAT + c) * NPAD + n];
    else if (c < KI) v = CT[(size_t)(bb * CAT + (c - CAT)) * NPAD + n];
    sx[idx] = v;
  }
  __syncthreads();
  const int lane = tid & 31;
  const int wave = tid >> 5;
  const int m = lane & 15;
  const int hi = lane >> 4;
  const int notiles = O / 16;            // 8 (gate) or 4 (update)
  const int ot = wave % notiles;
  const int g = wave / notiles;
  const int mtsPer = notiles / 2;        // M-tiles per wave (4 or 2)
  const int o = ot * 16 + m;
  const _Float16* wrow = Wt + ((size_t)n * O + o) * KPAD + (hi ? 8 : 0);
  for (int mi = 0; mi < mtsPer; ++mi) {
    const int mt = g * mtsPer + mi;
    v8f acc = {};
#pragma unroll
    for (int kc = 0; kc < KPAD / 32; ++kc) {  // 5 chunks
      const _Float16* pa = sx + (mt * 16 + m) * KPAD + kc * 32 + (hi ? 8 : 0);
      v8h a0 = *(const v8h*)pa;               // ds_load_b128
      v8h a1 = *(const v8h*)(pa + 16);
      v8h b0 = *(const v8h*)(wrow + kc * 32); // global b128
      v8h b1 = *(const v8h*)(wrow + kc * 32 + 16);
      v16h a, b;
#pragma unroll
      for (int e = 0; e < 8; ++e) {
        a[e] = a0[e]; a[e + 8] = a1[e];
        b[e] = b0[e]; b[e + 8] = b1[e];
      }
      acc = wmma_f16(a, b, acc);
    }
#pragma unroll
    for (int r = 0; r < 8; ++r) {
      const int bb = mt * 16 + r + (hi ? 8 : 0);  // batch index
      float v = acc[r] + bias[n * O + o];
      size_t base = (size_t)bb * NN + n;
      if (mode == 0) {                       // gate: sigmoid -> zr
        zr[base * OG + o] = (_Float16)(1.f / (1.f + __expf(-v)));
      } else {                               // candidate: tanh + combine
        float hc = tanhf(v);
        float rg = (float)zr[base * OG + (OU + o)];
        float st = state[base * OU + o];
        out[base * OU + o] = rg * st + (1.f - rg) * hc;
      }
    }
  }
}

extern "C" void kernel_launch(void* const* d_in, const int* in_sizes, int n_in,
                              void* d_out, int out_size, void* d_ws,
                              size_t ws_size, hipStream_t stream) {
  (void)in_sizes; (void)n_in; (void)out_size; (void)ws_size;
  const float* x   = (const float*)d_in[0];
  const float* st  = (const float*)d_in[1];
  const float* emb = (const float*)d_in[2];
  const float* gwp = (const float*)d_in[3];
  const float* gbp = (const float*)d_in[4];
  const float* uwp = (const float*)d_in[5];
  const float* ubp = (const float*)d_in[6];
  float* out = (float*)d_out;

  // workspace layout (~215 MB total)
  char* ws = (char*)d_ws;
  size_t off = 0;
  auto alloc = [&](size_t bytes) {
    size_t o = off;
    off += (bytes + 255) & ~(size_t)255;
    return o;
  };
  float*    logits = (float*)   (ws + alloc((size_t)NN * NN * 4));       // 16.0 MB
  _Float16* Ah     = (_Float16*)(ws + alloc((size_t)NN * NPAD * 2));     //  8.1 MB
  _Float16* XinT   = (_Float16*)(ws + alloc((size_t)COLS * NPAD * 2));   // 17.0 MB
  _Float16* CT     = (_Float16*)(ws + alloc((size_t)COLS * NPAD * 2));   // 17.0 MB
  _Float16* Wg     = (_Float16*)(ws + alloc((size_t)NN * OG * KPAD * 2));// 81.9 MB
  _Float16* Wu     = (_Float16*)(ws + alloc((size_t)NN * OU * KPAD * 2));// 41.0 MB
  float*    bg     = (float*)   (ws + alloc((size_t)NN * OG * 4));       //  1.0 MB
  float*    bu     = (float*)   (ws + alloc((size_t)NN * OU * 4));       //  0.5 MB
  _Float16* zr     = (_Float16*)(ws + alloc((size_t)BB * NN * OG * 2));  // 32.8 MB

  // adjacency
  logits_kernel<<<dim3(NN / 16, NN / 16), 32, 0, stream>>>(emb, logits);
  softmax_kernel<<<NN, 256, 0, stream>>>(logits, Ah);

  // per-node weights / biases (independent of activations)
  {
    size_t tg = (size_t)NN * OG * KPAD;
    node_weights_kernel<<<(unsigned)((tg + 255) / 256), 256, 0, stream>>>(emb, gwp, OG, Wg);
    size_t tu = (size_t)NN * OU * KPAD;
    node_weights_kernel<<<(unsigned)((tu + 255) / 256), 256, 0, stream>>>(emb, uwp, OU, Wu);
    node_bias_kernel<<<(NN * OG + 255) / 256, 256, 0, stream>>>(emb, gbp, OG, bg);
    node_bias_kernel<<<(NN * OU + 255) / 256, 256, 0, stream>>>(emb, ubp, OU, bu);
  }

  const size_t packTotal = (size_t)COLS * NPAD;
  const dim3 gemmGrid(25, COLS / 64);  // 125 M-tiles exactly, 66 col-blocks

  // gate pass
  pack_xin_kernel<<<(unsigned)((packTotal + 255) / 256), 256, 0, stream>>>(
      x, st, (const _Float16*)nullptr, 0, XinT);
  gemm_ax_kernel<<<gemmGrid, 160, 0, stream>>>(Ah, XinT, CT);
  node_gemm_kernel<<<NN, 256, 0, stream>>>(XinT, CT, Wg, bg, st, zr, out, OG, 0);

  // candidate pass + final combine
  pack_xin_kernel<<<(unsigned)((packTotal + 255) / 256), 256, 0, stream>>>(
      x, st, zr, 1, XinT);
  gemm_ax_kernel<<<gemmGrid, 160, 0, stream>>>(Ah, XinT, CT);
  node_gemm_kernel<<<NN, 256, 0, stream>>>(XinT, CT, Wu, bu, st, zr, out, OU, 1);
}
